// MultiHeadAttention_61160334295233
// MI455X (gfx1250) — matrix operand order repaired and checked by reading, not compiled, not run on hardware
//
#include <hip/hip_runtime.h>
#include <hip/hip_bf16.h>

// CDNA5 / gfx1250 wave32 WMMA attention pipeline.
typedef __attribute__((ext_vector_type(16))) _Float16 v16h;
typedef __attribute__((ext_vector_type(8)))  float    v8f;
typedef __attribute__((ext_vector_type(4)))  int      v4i;
typedef _Float16 half_t;

struct alignas(16) B16 { unsigned int u[4]; };

#if defined(__has_builtin)
#if __has_builtin(__builtin_amdgcn_global_load_async_to_lds_b128)
#define HAVE_ASYNC_LDS 1
#endif
#endif
#ifndef HAVE_ASYNC_LDS
#define HAVE_ASYNC_LDS 0
#endif

#if HAVE_ASYNC_LDS
using gv4i_ptr = __attribute__((address_space(1))) v4i*;  // global int4*
using lv4i_ptr = __attribute__((address_space(3))) v4i*;  // LDS int4*
#endif

__device__ __forceinline__ void wait_asynccnt0() {
#if defined(__has_builtin) && __has_builtin(__builtin_amdgcn_s_wait_asynccnt)
  __builtin_amdgcn_s_wait_asynccnt(0);
#else
  asm volatile("s_wait_asynccnt 0" ::: "memory");
#endif
}

// Assemble a 32-byte WMMA fragment from two 16-byte chunks.
__device__ __forceinline__ v16h load_frag16(const half_t* p0, const half_t* p1) {
  v16h r;
  ((B16*)&r)[0] = *(const B16*)p0;
  ((B16*)&r)[1] = *(const B16*)p1;
  return r;
}

// ---------------------------------------------------------------------------
// Elementwise conversions
// ---------------------------------------------------------------------------
__global__ void cvt_f16_kernel(const float* __restrict__ src, half_t* __restrict__ dst, size_t n) {
  size_t i = (size_t)blockIdx.x * blockDim.x + threadIdx.x;
  if (i < n) dst[i] = (half_t)src[i];
}

// W [K x N] f32 row-major  ->  WT [N x K] f16 row-major (transposed)
__global__ void wt_cvt_kernel(const float* __restrict__ w, half_t* __restrict__ wt,
                              int K, int N) {
  size_t i = (size_t)blockIdx.x * blockDim.x + threadIdx.x;
  size_t total = (size_t)K * N;
  if (i >= total) return;
  size_t k = i / N;
  size_t n = i % N;
  wt[n * (size_t)K + k] = (half_t)w[i];
}

// RoPE + repack: src [B*S, nheads*64] f32 -> dst [(b*nheads+h)*S + s]*64 + d  (f16)
__global__ void rope_pack_kernel(const float* __restrict__ src, half_t* __restrict__ dst,
                                 int nheads, float scale, int S, size_t total) {
  size_t idx = (size_t)blockIdx.x * blockDim.x + threadIdx.x;
  if (idx >= total) return;
  int i   = (int)(idx & 31);                 // rotation pair 0..31
  int h   = (int)((idx >> 5) % nheads);
  size_t tok = idx / ((size_t)nheads * 32);  // 0 .. B*S-1
  int s = (int)(tok % S);
  int b = (int)(tok / S);
  // inv_freq = ROPE_BASE^(-2i/64);  ln(10000)/64 = 0.14392747
  float inv = __expf(-(float)(2 * i) * 0.143927468f);
  float ang = (float)s * inv;
  float c = cosf(ang), sn = sinf(ang);
  size_t srow = tok * ((size_t)nheads * 64) + (size_t)h * 64 + 2 * i;
  float xe = src[srow], xo = src[srow + 1];
  float re = (xe * c - xo * sn) * scale;
  float ro = (xe * sn + xo * c) * scale;
  size_t o = (((size_t)b * nheads + h) * (size_t)S + s) * 64 + 2 * i;
  dst[o]     = (half_t)re;
  dst[o + 1] = (half_t)ro;
}

// V transpose pack: src [B*S, 512] f32 -> vt [(b*8+kvh)*64 + d]*S + s  (f16)
__global__ void v_pack_kernel(const float* __restrict__ src, half_t* __restrict__ dst,
                              int S, size_t total) {
  size_t idx = (size_t)blockIdx.x * blockDim.x + threadIdx.x;
  if (idx >= total) return;
  int c = (int)(idx % 512);
  size_t tok = idx / 512;
  int s = (int)(tok % S);
  int b = (int)(tok / S);
  int kvh = c >> 6, d = c & 63;
  dst[(((size_t)b * 8 + kvh) * 64 + d) * (size_t)S + s] = (half_t)src[idx];
}

// ---------------------------------------------------------------------------
// Tiled WMMA GEMM:  C[M,N](f32) = A[M,K](f16, row-major) * BT[N,K](f16)^T
// 128x128 block, 4 waves (2x2 of 64x64 wave tiles), K step 32, LDS double
// buffered with gfx1250 async global->LDS copies (ASYNCcnt) when available.
// ---------------------------------------------------------------------------
#define TBM 128
#define TBN 128
#define TBK 32
#define LDA 40   // LDS row stride in halves (padded: 80B rows, conflict-spread)

__global__ __launch_bounds__(128) void gemm_f16_kernel(
    const half_t* __restrict__ A, const half_t* __restrict__ BT,
    float* __restrict__ C, int M, int N, int K) {
  __shared__ __align__(16) half_t As[2][TBM * LDA];
  __shared__ __align__(16) half_t Bs[2][TBN * LDA];

  const int tid  = threadIdx.x;
  const int lane = tid & 31;
  const int wid  = tid >> 5;
  const int lh   = lane >> 4;   // half-wave (K/M group select)
  const int ll   = lane & 15;
  const size_t bm = (size_t)blockIdx.y * TBM;
  const size_t bn = (size_t)blockIdx.x * TBN;
  const int wm = (wid & 1) * 64;
  const int wn = (wid >> 1) * 64;

  const half_t* gArow = A  + (bm + tid) * (size_t)K;
  const half_t* gBrow = BT + (bn + tid) * (size_t)K;

  // Copy one 32-half row each of the A and B tiles for this thread.
  auto copy_tile = [&](int buf, int k0) {
    const half_t* gA = gArow + k0;
    const half_t* gB = gBrow + k0;
    half_t* lA = &As[buf][(size_t)tid * LDA];
    half_t* lB = &Bs[buf][(size_t)tid * LDA];
#if HAVE_ASYNC_LDS
    #pragma unroll
    for (int c = 0; c < 4; c++) {
      __builtin_amdgcn_global_load_async_to_lds_b128(
          (gv4i_ptr)(gA + c * 8), (lv4i_ptr)(lA + c * 8), 0, 0);
      __builtin_amdgcn_global_load_async_to_lds_b128(
          (gv4i_ptr)(gB + c * 8), (lv4i_ptr)(lB + c * 8), 0, 0);
    }
#else
    // Distinct temporaries so the 8 b128 loads pipeline before any ds_store.
    B16 ta0, ta1, ta2, ta3, tb0, tb1, tb2, tb3;
    const B16* ga = (const B16*)gA;
    const B16* gb = (const B16*)gB;
    ta0 = ga[0]; ta1 = ga[1]; ta2 = ga[2]; ta3 = ga[3];
    tb0 = gb[0]; tb1 = gb[1]; tb2 = gb[2]; tb3 = gb[3];
    B16* la = (B16*)lA;
    B16* lb = (B16*)lB;
    la[0] = ta0; la[1] = ta1; la[2] = ta2; la[3] = ta3;
    lb[0] = tb0; lb[1] = tb1; lb[2] = tb2; lb[3] = tb3;
#endif
  };

  v8f acc[4][4];
  #pragma unroll
  for (int i = 0; i < 4; i++)
    #pragma unroll
    for (int j = 0; j < 4; j++)
      #pragma unroll
      for (int e = 0; e < 8; e++) acc[i][j][e] = 0.0f;

  // Prologue: fill buffer 0.
  copy_tile(0, 0);
#if HAVE_ASYNC_LDS
  wait_asynccnt0();
#endif
  __syncthreads();

  int cur = 0;
  for (int k0 = 0; k0 < K; k0 += TBK) {
    const bool more = (k0 + TBK < K);
    if (more) copy_tile(cur ^ 1, k0 + TBK);  // overlap next-tile DMA with math
#if !HAVE_ASYNC_LDS
    if (more) {
      __builtin_prefetch(gArow + k0 + 2 * TBK, 0, 1);
      __builtin_prefetch(gBrow + k0 + 2 * TBK, 0, 1);
    }
#endif

    const half_t* Ab = As[cur];
    const half_t* Bb = Bs[cur];
    v16h af[4], bf[4];
    #pragma unroll
    for (int i = 0; i < 4; i++) {
      // A-frag 16x32: lane holds row M=ll, K = lh*16 + 0..15 contiguous
      const half_t* p = Ab + (size_t)(wm + 16 * i + ll) * LDA + lh * 16;
      af[i] = load_frag16(p, p + 8);
    }
    #pragma unroll
    for (int j = 0; j < 4; j++) {
      // B-frag 32x16: lane holds column N=ll, K = lh*16 + 0..15 contiguous (BT row)
      const half_t* p = Bb + (size_t)(wn + 16 * j + ll) * LDA + lh * 16;
      bf[j] = load_frag16(p, p + 8);
    }
    #pragma unroll
    for (int i = 0; i < 4; i++)
      #pragma unroll
      for (int j = 0; j < 4; j++)
        acc[i][j] = __builtin_amdgcn_wmma_f32_16x16x32_f16(
            false, af[i], false, bf[j], (short)0, acc[i][j], false, false);

#if HAVE_ASYNC_LDS
    if (more) wait_asynccnt0();
#endif
    __syncthreads();
    cur ^= 1;
  }

  // C layout: VGPR r, lanes 0-15: M=r; lanes 16-31: M=r+8; N = ll
  #pragma unroll
  for (int i = 0; i < 4; i++)
    #pragma unroll
    for (int j = 0; j < 4; j++) {
      const size_t mrow = bm + wm + 16 * i + lh * 8;
      const size_t ncol = bn + wn + 16 * j + ll;
      float* cp = C + mrow * (size_t)N + ncol;
      #pragma unroll
      for (int r = 0; r < 8; r++) cp[(size_t)r * N] = acc[i][j][r];
    }
}

// ---------------------------------------------------------------------------
// Flash attention: one wave per (batch*head, 16-row q block).
// qh [bh][S][64] f16 (pre-scaled by 1/8), kh [b*KVH+kvh][S][64] f16,
// vt [b*KVH+kvh][64][S] f16, ah [B*S][2048] f16 (token-major, col = h*64+d).
// ---------------------------------------------------------------------------
__global__ __launch_bounds__(32) void attn_kernel(
    const half_t* __restrict__ qh, const half_t* __restrict__ kh,
    const half_t* __restrict__ vt, half_t* __restrict__ ah, int S) {
  __shared__ __align__(16) half_t Ps[16 * LDA];   // 16x32 prob tile, padded

  const int lane = threadIdx.x & 31;
  const int lh = lane >> 4, ll = lane & 15;
  const int qblk = blockIdx.x;          // S/16 blocks
  const int bh = blockIdx.y;            // b*32 + h
  const int b = bh >> 5, h = bh & 31;
  const int kvh = h >> 2;               // GQA: 4 query heads per KV head
  const int D = 64;

  const half_t* qbase = qh + ((size_t)bh * S + (size_t)qblk * 16) * D;
  const half_t* kbase = kh + ((size_t)(b * 8 + kvh) * S) * D;
  const half_t* vbase = vt + ((size_t)(b * 8 + kvh) * D) * S;

  v16h aq[2];
  #pragma unroll
  for (int ks = 0; ks < 2; ks++) {
    const half_t* p = qbase + (size_t)ll * D + ks * 32 + lh * 16;
    aq[ks] = load_frag16(p, p + 8);
  }

  v8f acc[4];
  #pragma unroll
  for (int dt = 0; dt < 4; dt++)
    #pragma unroll
    for (int e = 0; e < 8; e++) acc[dt][e] = 0.0f;
  float mrow[8], lrow[8];
  #pragma unroll
  for (int r = 0; r < 8; r++) { mrow[r] = -1e30f; lrow[r] = 0.0f; }

  const int qmax = qblk * 16 + 15;
  for (int c0 = 0; c0 <= qmax; c0 += 32) {
    // scores for 32 K columns (two 16x16 WMMA tiles, K-dim 64 = 2 steps)
    v8f sc0, sc1;
    #pragma unroll
    for (int e = 0; e < 8; e++) { sc0[e] = 0.0f; sc1[e] = 0.0f; }
    #pragma unroll
    for (int ks = 0; ks < 2; ks++) {
      const half_t* kp0 = kbase + (size_t)(c0 + ll) * D + ks * 32 + lh * 16;
      const half_t* kp1 = kbase + (size_t)(c0 + 16 + ll) * D + ks * 32 + lh * 16;
      v16h bk0 = load_frag16(kp0, kp0 + 8);
      v16h bk1 = load_frag16(kp1, kp1 + 8);
      sc0 = __builtin_amdgcn_wmma_f32_16x16x32_f16(false, aq[ks], false, bk0, (short)0, sc0, false, false);
      sc1 = __builtin_amdgcn_wmma_f32_16x16x32_f16(false, aq[ks], false, bk1, (short)0, sc1, false, false);
    }
    // causal mask + online softmax (row = r + 8*lh, col group N = ll)
    #pragma unroll
    for (int r = 0; r < 8; r++) {
      const int mg = qblk * 16 + r + lh * 8;
      if (c0 + ll      > mg) sc0[r] = -3.0e38f;
      if (c0 + 16 + ll > mg) sc1[r] = -3.0e38f;
      float v = fmaxf(sc0[r], sc1[r]);
      v = fmaxf(v, __shfl_xor(v, 1, 32));
      v = fmaxf(v, __shfl_xor(v, 2, 32));
      v = fmaxf(v, __shfl_xor(v, 4, 32));
      v = fmaxf(v, __shfl_xor(v, 8, 32));
      const float mn = fmaxf(mrow[r], v);
      const float alpha = __expf(mrow[r] - mn);
      const float p0 = __expf(sc0[r] - mn);
      const float p1 = __expf(sc1[r] - mn);
      sc0[r] = p0; sc1[r] = p1;
      float ssum = p0 + p1;
      ssum += __shfl_xor(ssum, 1, 32);
      ssum += __shfl_xor(ssum, 2, 32);
      ssum += __shfl_xor(ssum, 4, 32);
      ssum += __shfl_xor(ssum, 8, 32);
      lrow[r] = lrow[r] * alpha + ssum;
      mrow[r] = mn;
      #pragma unroll
      for (int dt = 0; dt < 4; dt++) acc[dt][r] *= alpha;
    }
    // C-layout -> A-layout re-tile through LDS (single wave: barrier is a fence)
    #pragma unroll
    for (int r = 0; r < 8; r++) {
      const int m = r + lh * 8;
      Ps[m * LDA + ll]      = (half_t)sc0[r];
      Ps[m * LDA + 16 + ll] = (half_t)sc1[r];
    }
    __syncthreads();
    v16h ap;
    {
      const half_t* pp = Ps + (size_t)ll * LDA + lh * 16;
      ap = load_frag16(pp, pp + 8);
    }
    // out += P(16x32) * V(32x64): 4 N-tiles of d, V^T rows give contiguous frags
    #pragma unroll
    for (int dt = 0; dt < 4; dt++) {
      const half_t* vp = vbase + (size_t)(dt * 16 + ll) * S + c0 + lh * 16;
      v16h bv = load_frag16(vp, vp + 8);
      acc[dt] = __builtin_amdgcn_wmma_f32_16x16x32_f16(false, ap, false, bv, (short)0, acc[dt], false, false);
    }
    __syncthreads();
  }

  // normalize + store token-major f16 for the output projection
  #pragma unroll
  for (int dt = 0; dt < 4; dt++)
    #pragma unroll
    for (int r = 0; r < 8; r++) {
      const int sq = qblk * 16 + r + lh * 8;
      const float o = acc[dt][r] / lrow[r];
      ah[((size_t)b * S + sq) * 2048 + h * 64 + dt * 16 + ll] = (half_t)o;
    }
}

// ---------------------------------------------------------------------------
// Launcher
// ---------------------------------------------------------------------------
extern "C" void kernel_launch(void* const* d_in, const int* in_sizes, int n_in,
                              void* d_out, int out_size, void* d_ws, size_t ws_size,
                              hipStream_t stream) {
  (void)in_sizes; (void)n_in; (void)out_size; (void)ws_size;
  const float* x  = (const float*)d_in[0];
  const float* wq = (const float*)d_in[1];
  const float* wk = (const float*)d_in[2];
  const float* wv = (const float*)d_in[3];
  const float* wo = (const float*)d_in[4];
  float* out = (float*)d_out;

  const int B = 2, S = 2048, DIM = 2048, H = 32, KVH = 8, D = 64;
  const size_t M = (size_t)B * S;  // 4096 tokens

  char* p = (char*)d_ws;
  auto take = [&](size_t bytes) -> char* {
    char* q = p;
    p += (bytes + 255) & ~(size_t)255;
    return q;
  };
  half_t* xh  = (half_t*)take(M * DIM * 2);
  half_t* wqT = (half_t*)take((size_t)DIM * DIM * 2);
  half_t* wkT = (half_t*)take((size_t)512 * DIM * 2);
  half_t* wvT = (half_t*)take((size_t)512 * DIM * 2);
  half_t* woT = (half_t*)take((size_t)DIM * DIM * 2);
  float*  q32 = (float*)take(M * DIM * 4);
  float*  k32 = (float*)take(M * 512 * 4);
  float*  v32 = (float*)take(M * 512 * 4);
  half_t* qh  = (half_t*)take(M * DIM * 2);
  half_t* kh  = (half_t*)take((size_t)B * KVH * S * D * 2);
  half_t* vt  = (half_t*)take((size_t)B * KVH * S * D * 2);
  half_t* ah  = (half_t*)take(M * DIM * 2);

  // 1) f32 -> f16 conversions (weights transposed so WMMA B-frags are contiguous)
  { size_t n = M * DIM;
    cvt_f16_kernel<<<(unsigned)((n + 255) / 256), 256, 0, stream>>>(x, xh, n); }
  { size_t n = (size_t)DIM * DIM;
    wt_cvt_kernel<<<(unsigned)((n + 255) / 256), 256, 0, stream>>>(wq, wqT, DIM, DIM); }
  { size_t n = (size_t)DIM * 512;
    wt_cvt_kernel<<<(unsigned)((n + 255) / 256), 256, 0, stream>>>(wk, wkT, DIM, 512); }
  { size_t n = (size_t)DIM * 512;
    wt_cvt_kernel<<<(unsigned)((n + 255) / 256), 256, 0, stream>>>(wv, wvT, DIM, 512); }
  { size_t n = (size_t)DIM * DIM;
    wt_cvt_kernel<<<(unsigned)((n + 255) / 256), 256, 0, stream>>>(wo, woT, DIM, DIM); }

  // 2) QKV projections (f16 WMMA, f32 accumulate)
  gemm_f16_kernel<<<dim3(DIM / TBN, (unsigned)(M / TBM)), 128, 0, stream>>>(xh, wqT, q32, (int)M, DIM, DIM);
  gemm_f16_kernel<<<dim3(512 / TBN, (unsigned)(M / TBM)), 128, 0, stream>>>(xh, wkT, k32, (int)M, 512, DIM);
  gemm_f16_kernel<<<dim3(512 / TBN, (unsigned)(M / TBM)), 128, 0, stream>>>(xh, wvT, v32, (int)M, 512, DIM);

  // 3) RoPE (+1/sqrt(64) folded into q) and attention-friendly repacks
  { size_t n = M * (size_t)H * 32;
    rope_pack_kernel<<<(unsigned)((n + 255) / 256), 256, 0, stream>>>(q32, qh, H, 0.125f, S, n); }
  { size_t n = M * (size_t)KVH * 32;
    rope_pack_kernel<<<(unsigned)((n + 255) / 256), 256, 0, stream>>>(k32, kh, KVH, 1.0f, S, n); }
  { size_t n = M * 512;
    v_pack_kernel<<<(unsigned)((n + 255) / 256), 256, 0, stream>>>(v32, vt, S, n); }

  // 4) causal flash attention (one wave per 16-row q block)
  attn_kernel<<<dim3(S / 16, B * H), 32, 0, stream>>>(qh, kh, vt, ah, S);

  // 5) output projection -> f32 result
  gemm_f16_kernel<<<dim3(DIM / TBN, (unsigned)(M / TBM)), 128, 0, stream>>>(ah, woT, out, (int)M, DIM, DIM);
}
